// DiTBlock_8040178778722
// MI455X (gfx1250) — matrix operand order repaired and checked by reading, not compiled, not run on hardware
//
#include <hip/hip_runtime.h>
#include <hip/hip_bf16.h>
#include <math.h>

// ---------------------------------------------------------------------------
// DiT block for MI455X (gfx1250, wave32, WMMA).
// All large GEMMs run through v_wmma_f32_16x16x32_bf16 (fp32 accumulate).
// Wave tile 64x64 (16 WMMA per 16 b128 loads); block tile 256x128 (8 waves).
// ---------------------------------------------------------------------------

#define B_  2
#define T_  2048
#define F_  1024
#define H_  16
#define D_  64
#define M_  4
#define C_  512
#define HD_ (H_ * D_)      // 1024
#define KV_ (2 * D_)       // 128
#define MF_ (M_ * F_)      // 4096

typedef __attribute__((ext_vector_type(16))) __bf16          v16bf;
typedef __attribute__((ext_vector_type(16))) unsigned short  v16u;
typedef __attribute__((ext_vector_type(8)))  unsigned short  v8u;
typedef __attribute__((ext_vector_type(8)))  float           v8f;

#define DEV static __device__ __forceinline__

DEV unsigned short f2b(float f) {          // fp32 -> bf16, round-nearest-even
    unsigned u = __float_as_uint(f);
    u += 0x7FFFu + ((u >> 16) & 1u);
    return (unsigned short)(u >> 16);
}

union FragBF { v16u u; v16bf b; };

// Build a 16-element bf16 fragment from two contiguous 8x16-bit chunks.
DEV v16bf make_frag(const unsigned short* p0, const unsigned short* p1) {
    v8u lo = *(const v8u*)p0;
    v8u hi = *(const v8u*)p1;
    FragBF f;
#pragma unroll
    for (int i = 0; i < 8; ++i) { f.u[i] = lo[i]; f.u[i + 8] = hi[i]; }
    return f.b;
}

// A fragment (16x32, row-major A): lane<16 holds K {k0..k0+7, k0+16..k0+23},
// lane>=16 holds K {k0+8..k0+15, k0+24..k0+31} of row m0+(lane&15).
DEV v16bf load_a_frag(const unsigned short* rowptr /* = A + row*lda + k0 + ka */) {
    return make_frag(rowptr, rowptr + 16);
}
// B fragment from pre-transposed BT (N x K): lane<16 holds K {k0..k0+7, k0+16..k0+23} of
// col n0+(lane&15); lane>=16 holds K {k0+8..k0+15, k0+24..k0+31}.
DEV v16bf load_b_frag(const unsigned short* rowptr /* = BT + col*ldbt + k0 + kb */) {
    return make_frag(rowptr, rowptr + 16);
}

// ---------------------------------------------------------------------------
// Generic bf16 WMMA GEMM:  C[M,N] = A[M,K] (bf16) * BT[N,K]^T (bf16)
// Block = 256 threads = 8 waves (4 rows x 2 cols); wave tile 64x64.
// Block tile: 256 (M) x 128 (N).
// ---------------------------------------------------------------------------
template <bool BF16OUT, bool BIAS, bool RES, bool GELUF>
__global__ __launch_bounds__(256) void gemm_wmma(
    const unsigned short* __restrict__ A, const unsigned short* __restrict__ BT,
    void* __restrict__ Cout, const float* __restrict__ bias,
    const float* __restrict__ Res, int M, int N, int K,
    int lda, int ldbt, int ldc)
{
    const int wid  = threadIdx.x >> 5;
    const int lane = threadIdx.x & 31;
    const int wr   = wid >> 1;                 // 0..3 (M direction)
    const int wc   = wid & 1;                  // 0..1 (N direction)
    const int m0   = blockIdx.y * 256 + wr * 64;
    const int n0   = blockIdx.x * 128 + wc * 64;
    if (m0 >= M || n0 >= N) return;            // wave-uniform

    v8f acc[4][4] = {};

    const int arow = m0 + (lane & 15);
    const int bcol = n0 + (lane & 15);
    const int ka   = (lane >> 4) << 3;         // 0 or 8
    const int kb   = (lane >> 4) << 3;         // 0 or 8

    const unsigned short* Ab[4];
    const unsigned short* Bb[4];
#pragma unroll
    for (int i = 0; i < 4; ++i) Ab[i] = A  + (size_t)(arow + 16 * i) * lda  + ka;
#pragma unroll
    for (int j = 0; j < 4; ++j) Bb[j] = BT + (size_t)(bcol + 16 * j) * ldbt + kb;

    for (int k0 = 0; k0 < K; k0 += 32) {
        v16bf af[4], bf[4];
#pragma unroll
        for (int i = 0; i < 4; ++i) af[i] = load_a_frag(Ab[i] + k0);
#pragma unroll
        for (int j = 0; j < 4; ++j) bf[j] = load_b_frag(Bb[j] + k0);
#pragma unroll
        for (int i = 0; i < 4; ++i)
#pragma unroll
            for (int j = 0; j < 4; ++j)
                acc[i][j] = __builtin_amdgcn_wmma_f32_16x16x32_bf16(
                    false, af[i], false, bf[j], (short)0, acc[i][j], false, false);
    }

    // C layout: VGPR r of lane l -> M = r + ((l>=16)?8:0), N = l&15 (per frag).
    const int cn = n0 + (lane & 15);
    const int rb = (lane >> 4) << 3;
#pragma unroll
    for (int i = 0; i < 4; ++i) {
#pragma unroll
        for (int r = 0; r < 8; ++r) {
            const int cm = m0 + i * 16 + rb + r;
#pragma unroll
            for (int j = 0; j < 4; ++j) {
                float v = acc[i][j][r];
                const int cc = cn + j * 16;
                if (BIAS)  v += bias[cc];
                if (RES)   v += Res[(size_t)cm * ldc + cc];
                if (GELUF) v = 0.5f * v * (1.0f + erff(v * 0.70710678118654752f));
                if (BF16OUT) ((unsigned short*)Cout)[(size_t)cm * ldc + cc] = f2b(v);
                else         ((float*)Cout)[(size_t)cm * ldc + cc] = v;
            }
        }
    }
}

// ---------------------------------------------------------------------------
// Transpose + convert: in fp32 [K x N] row-major -> out bf16 [N x K] row-major
// ---------------------------------------------------------------------------
__global__ __launch_bounds__(256) void transpose_f32_bf16(
    const float* __restrict__ in, unsigned short* __restrict__ out, int K, int N)
{
    __shared__ float tile[32][33];
    const int k0 = blockIdx.x * 32, n0 = blockIdx.y * 32;
    const int tx = threadIdx.x, ty = threadIdx.y;   // block (32,8)
#pragma unroll
    for (int j = 0; j < 4; ++j)
        tile[ty + j * 8][tx] = in[(size_t)(k0 + ty + j * 8) * N + n0 + tx];
    __syncthreads();
#pragma unroll
    for (int j = 0; j < 4; ++j)
        out[(size_t)(n0 + ty + j * 8) * K + k0 + tx] = f2b(tile[tx][ty + j * 8]);
}

// v^T:  kvbf [B*T x 2D] cols D..2D-1  ->  vT [B][D x T] bf16
__global__ __launch_bounds__(256) void vtranspose_kernel(
    const unsigned short* __restrict__ kvbf, unsigned short* __restrict__ vT)
{
    int idx = blockIdx.x * 256 + threadIdx.x;        // over B*T*D
    if (idx >= B_ * T_ * D_) return;
    int b = idx / (T_ * D_);
    int r = idx % (T_ * D_);
    int t = r / D_, d = r % D_;
    vT[((size_t)b * D_ + d) * T_ + t] = kvbf[((size_t)(b * T_ + t)) * KV_ + D_ + d];
}

// ---------------------------------------------------------------------------
// cond @ W + bias  (M = B = 2: trivial, do on VALU)
// ---------------------------------------------------------------------------
__global__ __launch_bounds__(256) void cond_mod_kernel(
    const float* __restrict__ cond, const float* __restrict__ W,
    const float* __restrict__ bias, float* __restrict__ out)
{
    int idx = blockIdx.x * 256 + threadIdx.x;        // over B*F
    int b = idx / F_, f = idx % F_;
    float acc = bias[f];
    for (int c = 0; c < C_; ++c)
        acc += cond[b * C_ + c] * W[(size_t)c * F_ + f];
    out[idx] = acc;
}

// ---------------------------------------------------------------------------
// LayerNorm / AdaLN: one row (F=1024) per 256-thread block.
// ---------------------------------------------------------------------------
template <bool MOD, bool BF16OUT>
__global__ __launch_bounds__(256) void ln_kernel(
    const float* __restrict__ X, const float* __restrict__ g,
    const float* __restrict__ be, const float* __restrict__ nw,
    const float* __restrict__ nb, void* __restrict__ out)
{
    __shared__ float sh[256];
    const int row = blockIdx.x;                      // B*T rows
    const int b   = row / T_;
    const int tid = threadIdx.x;
    const float* x = X + (size_t)row * F_;

    float v[4], s = 0.f, sq = 0.f;
#pragma unroll
    for (int i = 0; i < 4; ++i) {
        float t = x[tid + i * 256];
        v[i] = t; s += t; sq += t * t;
    }
    sh[tid] = s; __syncthreads();
    for (int st = 128; st > 0; st >>= 1) { if (tid < st) sh[tid] += sh[tid + st]; __syncthreads(); }
    s = sh[0]; __syncthreads();
    sh[tid] = sq; __syncthreads();
    for (int st = 128; st > 0; st >>= 1) { if (tid < st) sh[tid] += sh[tid + st]; __syncthreads(); }
    sq = sh[0];

    const float mu   = s * (1.0f / F_);
    const float var  = sq * (1.0f / F_) - mu * mu;
    const float rstd = rsqrtf(var + 1e-5f);
#pragma unroll
    for (int i = 0; i < 4; ++i) {
        const int f = tid + i * 256;
        float y = (v[i] - mu) * rstd * nw[f] + nb[f];
        if (MOD) y = y * (1.0f + g[(size_t)b * F_ + f]) + be[(size_t)b * F_ + f];
        if (BF16OUT) ((unsigned short*)out)[(size_t)row * F_ + f] = f2b(y);
        else         ((float*)out)[(size_t)row * F_ + f] = y;
    }
}

// ---------------------------------------------------------------------------
// Row softmax (n = 2048) with scale; fp32 in, bf16 out.
// ---------------------------------------------------------------------------
__global__ __launch_bounds__(256) void softmax_kernel(
    const float* __restrict__ S, unsigned short* __restrict__ P, float scale)
{
    __shared__ float sh[256];
    const int row = blockIdx.x;
    const int tid = threadIdx.x;
    const float* s = S + (size_t)row * T_;
    unsigned short* p = P + (size_t)row * T_;

    float v[8], m = -3.4e38f;
#pragma unroll
    for (int i = 0; i < 8; ++i) { v[i] = s[tid + i * 256] * scale; m = fmaxf(m, v[i]); }
    sh[tid] = m; __syncthreads();
    for (int st = 128; st > 0; st >>= 1) { if (tid < st) sh[tid] = fmaxf(sh[tid], sh[tid + st]); __syncthreads(); }
    m = sh[0]; __syncthreads();
    float sum = 0.f;
#pragma unroll
    for (int i = 0; i < 8; ++i) { v[i] = __expf(v[i] - m); sum += v[i]; }
    sh[tid] = sum; __syncthreads();
    for (int st = 128; st > 0; st >>= 1) { if (tid < st) sh[tid] += sh[tid + st]; __syncthreads(); }
    const float inv = 1.0f / sh[0];
#pragma unroll
    for (int i = 0; i < 8; ++i) p[tid + i * 256] = f2b(v[i] * inv);
}

// ---------------------------------------------------------------------------
// Host orchestration
// ---------------------------------------------------------------------------
extern "C" void kernel_launch(void* const* d_in, const int* in_sizes, int n_in,
                              void* d_out, int out_size, void* d_ws, size_t ws_size,
                              hipStream_t stream)
{
    (void)in_sizes; (void)n_in; (void)out_size; (void)ws_size;

    const float* x       = (const float*)d_in[0];
    const float* cond    = (const float*)d_in[1];
    const float* amod_nw = (const float*)d_in[2];
    const float* amod_nb = (const float*)d_in[3];
    const float* amod_gw = (const float*)d_in[4];
    const float* amod_gb = (const float*)d_in[5];
    const float* amod_bw = (const float*)d_in[6];
    const float* amod_bb = (const float*)d_in[7];
    const float* attn_nw = (const float*)d_in[8];
    const float* attn_nb = (const float*)d_in[9];
    const float* wq      = (const float*)d_in[10];
    const float* wkv     = (const float*)d_in[11];
    const float* wo      = (const float*)d_in[12];
    const float* wo_b    = (const float*)d_in[13];
    const float* fmod_nw = (const float*)d_in[14];
    const float* fmod_nb = (const float*)d_in[15];
    const float* fmod_gw = (const float*)d_in[16];
    const float* fmod_gb = (const float*)d_in[17];
    const float* fmod_bw = (const float*)d_in[18];
    const float* fmod_bb = (const float*)d_in[19];
    const float* w1      = (const float*)d_in[20];
    const float* b1      = (const float*)d_in[21];
    const float* w2      = (const float*)d_in[22];
    const float* b2      = (const float*)d_in[23];

    // ---- workspace carve ----
    char* p = (char*)d_ws;
    auto alloc = [&](size_t bytes) -> char* {
        char* r = p; p += (bytes + 255) & ~(size_t)255; return r;
    };
    const size_t BT = (size_t)B_ * T_;
    float*          g_a   = (float*)alloc(B_ * F_ * 4);
    float*          be_a  = (float*)alloc(B_ * F_ * 4);
    float*          g_f   = (float*)alloc(B_ * F_ * 4);
    float*          be_f  = (float*)alloc(B_ * F_ * 4);
    float*          h     = (float*)alloc(BT * F_ * 4);
    float*          x1    = (float*)alloc(BT * F_ * 4);
    float*          scores= (float*)alloc((size_t)T_ * T_ * 4);
    unsigned short* hn    = (unsigned short*)alloc(BT * F_ * 2);
    unsigned short* qbf   = (unsigned short*)alloc(BT * HD_ * 2);
    unsigned short* kvbf  = (unsigned short*)alloc(BT * KV_ * 2);
    unsigned short* vT    = (unsigned short*)alloc((size_t)B_ * D_ * T_ * 2);
    unsigned short* probs = (unsigned short*)alloc((size_t)T_ * T_ * 2);
    unsigned short* obf   = (unsigned short*)alloc(BT * HD_ * 2);
    unsigned short* h2    = (unsigned short*)alloc(BT * F_ * 2);
    unsigned short* midbf = (unsigned short*)alloc(BT * MF_ * 2);
    unsigned short* wq_t  = (unsigned short*)alloc((size_t)HD_ * F_ * 2);
    unsigned short* wkv_t = (unsigned short*)alloc((size_t)KV_ * F_ * 2);
    unsigned short* wo_t  = (unsigned short*)alloc((size_t)F_ * HD_ * 2);
    unsigned short* w1_t  = (unsigned short*)alloc((size_t)MF_ * F_ * 2);
    unsigned short* w2_t  = (unsigned short*)alloc((size_t)F_ * MF_ * 2);

    const dim3 tb(32, 8);
    // ---- weights: fp32 [K x N] -> bf16 [N x K] ----
    transpose_f32_bf16<<<dim3(F_ / 32, HD_ / 32), tb, 0, stream>>>(wq,  wq_t,  F_,  HD_);
    transpose_f32_bf16<<<dim3(F_ / 32, KV_ / 32), tb, 0, stream>>>(wkv, wkv_t, F_,  KV_);
    transpose_f32_bf16<<<dim3(HD_ / 32, F_ / 32), tb, 0, stream>>>(wo,  wo_t,  HD_, F_);
    transpose_f32_bf16<<<dim3(F_ / 32, MF_ / 32), tb, 0, stream>>>(w1,  w1_t,  F_,  MF_);
    transpose_f32_bf16<<<dim3(MF_ / 32, F_ / 32), tb, 0, stream>>>(w2,  w2_t,  MF_, F_);

    // ---- conditioning modulation (tiny M=2 GEMMs) ----
    cond_mod_kernel<<<(B_ * F_) / 256, 256, 0, stream>>>(cond, amod_gw, amod_gb, g_a);
    cond_mod_kernel<<<(B_ * F_) / 256, 256, 0, stream>>>(cond, amod_bw, amod_bb, be_a);
    cond_mod_kernel<<<(B_ * F_) / 256, 256, 0, stream>>>(cond, fmod_gw, fmod_gb, g_f);
    cond_mod_kernel<<<(B_ * F_) / 256, 256, 0, stream>>>(cond, fmod_bw, fmod_bb, be_f);

    // ---- h = adaln(x); hn = ln(h) in bf16 ----
    ln_kernel<true,  false><<<BT, 256, 0, stream>>>(x, g_a, be_a, amod_nw, amod_nb, h);
    ln_kernel<false, true ><<<BT, 256, 0, stream>>>(h, nullptr, nullptr, attn_nw, attn_nb, hn);

    // ---- q / kv projections ----
    gemm_wmma<true, false, false, false><<<dim3(HD_ / 128, BT / 256), 256, 0, stream>>>(
        hn, wq_t, qbf, nullptr, nullptr, (int)BT, HD_, F_, F_, F_, HD_);
    gemm_wmma<true, false, false, false><<<dim3(KV_ / 128, BT / 256), 256, 0, stream>>>(
        hn, wkv_t, kvbf, nullptr, nullptr, (int)BT, KV_, F_, F_, F_, KV_);
    vtranspose_kernel<<<(B_ * T_ * D_) / 256, 256, 0, stream>>>(kvbf, vT);

    // ---- attention: static 32-step (b,h) loop reusing the scores buffer ----
    const float scale = 0.125f;   // 1/sqrt(64)
    for (int b = 0; b < B_; ++b) {
        for (int hh = 0; hh < H_; ++hh) {
            const unsigned short* qh = qbf  + (size_t)b * T_ * HD_ + hh * D_;
            const unsigned short* kb = kvbf + (size_t)b * T_ * KV_;          // k cols 0..D-1
            const unsigned short* vb = vT   + (size_t)b * D_ * T_;
            unsigned short*       oh = obf  + (size_t)b * T_ * HD_ + hh * D_;
            // scores[T,T] = q[b,:,h,:] @ k[b]^T
            gemm_wmma<false, false, false, false><<<dim3(T_ / 128, T_ / 256), 256, 0, stream>>>(
                qh, kb, scores, nullptr, nullptr, T_, T_, D_, HD_, KV_, T_);
            softmax_kernel<<<T_, 256, 0, stream>>>(scores, probs, scale);
            // o[b,:,h,:] = P @ v[b]  (N=64: wave col 1 exits, col 0 covers it)
            gemm_wmma<true, false, false, false><<<dim3(1, T_ / 256), 256, 0, stream>>>(
                probs, vb, oh, nullptr, nullptr, T_, D_, T_, T_, T_, HD_);
        }
    }

    // ---- x1 = o @ wo + wo_b + h ----
    gemm_wmma<false, true, true, false><<<dim3(F_ / 128, BT / 256), 256, 0, stream>>>(
        obf, wo_t, x1, wo_b, h, (int)BT, F_, HD_, HD_, HD_, F_);

    // ---- h2 = adaln(x1) in bf16 ----
    ln_kernel<true, true><<<BT, 256, 0, stream>>>(x1, g_f, be_f, fmod_nw, fmod_nb, h2);

    // ---- FFN: mid = gelu(h2 @ w1 + b1) ; out = mid @ w2 + b2 + x1 ----
    gemm_wmma<true, true, false, true><<<dim3(MF_ / 128, BT / 256), 256, 0, stream>>>(
        h2, w1_t, midbf, b1, nullptr, (int)BT, MF_, F_, F_, F_, MF_);
    gemm_wmma<false, true, true, false><<<dim3(F_ / 128, BT / 256), 256, 0, stream>>>(
        midbf, w2_t, (float*)d_out, b2, x1, (int)BT, F_, MF_, MF_, MF_, F_);
}
